// ResidualBlock_88725434401178
// MI455X (gfx1250) — matrix operand order repaired and checked by reading, not compiled, not run on hardware
//
#include <hip/hip_runtime.h>
#include <hip/hip_bf16.h>

#define V_NODES 100000
#define NNZ_E   1600000

typedef __attribute__((ext_vector_type(16))) __bf16 v16bf;
typedef __attribute__((ext_vector_type(8)))  float  v8f;

// ---------------------------------------------------------------- utilities
__global__ __launch_bounds__(256) void fill_zero_kernel(float* __restrict__ p, long n) {
    long i = (long)blockIdx.x * blockDim.x + threadIdx.x;
    long s = (long)gridDim.x * blockDim.x;
    for (; i < n; i += s) p[i] = 0.0f;
}

__global__ __launch_bounds__(256) void neg_copy_kernel(float* __restrict__ dst,
                                                       const float* __restrict__ src, long n) {
    long i = (long)blockIdx.x * blockDim.x + threadIdx.x;
    long s = (long)gridDim.x * blockDim.x;
    for (; i < n; i += s) dst[i] = -src[i];
}

// (256, V) -> (V, 256) tiled transpose: x(B,Cin,V) -> x0(V, B*Cin)
__global__ __launch_bounds__(256) void transpose_kernel(const float* __restrict__ in,
                                                        float* __restrict__ out) {
    __shared__ float tile[32][33];
    int vb = blockIdx.x * 32;     // V block (V = 3125*32 exactly)
    int rb = blockIdx.y * 32;     // row block (256 = 8*32 exactly)
    int tx = threadIdx.x, ty = threadIdx.y;   // (32,8)
    #pragma unroll
    for (int j = 0; j < 32; j += 8)
        tile[ty + j][tx] = in[(long)(rb + ty + j) * V_NODES + vb + tx];
    __syncthreads();
    #pragma unroll
    for (int j = 0; j < 32; j += 8)
        out[(long)(vb + ty + j) * 256 + rb + tx] = tile[tx][ty + j];
}

// Edge-parallel SpMM scatter: acc[row,:] += coef * val * y[col,:]
__global__ __launch_bounds__(256) void spmm_scatter(const int* __restrict__ rows,
                                                    const int* __restrict__ cols,
                                                    const float* __restrict__ vals,
                                                    const float* __restrict__ y,
                                                    float* __restrict__ acc,
                                                    int nnz, int C, float coef) {
    for (int e = blockIdx.x; e < nnz; e += gridDim.x) {
        int r = rows[e];
        int c = cols[e];
        float v = coef * vals[e];
        const float* src = y + (long)c * C;
        float* dst = acc + (long)r * C;
        for (int t = threadIdx.x; t < C; t += 256)
            atomicAdd(&dst[t], v * src[t]);
    }
}

// ------------------------------------------------------------ BatchNorm
__global__ __launch_bounds__(256) void bn_stats_kernel(const float* __restrict__ h,
                                                       float* __restrict__ sums,
                                                       float* __restrict__ sumsq, long n) {
    long g = (long)blockIdx.x * blockDim.x + threadIdx.x;
    long T = (long)gridDim.x * blockDim.x;      // multiple of 128 -> channel is fixed per thread
    float s = 0.0f, s2 = 0.0f;
    for (long i = g; i < n; i += T) { float x = h[i]; s += x; s2 += x * x; }
    int c = (int)(g & 127);
    atomicAdd(&sums[c], s);
    atomicAdd(&sumsq[c], s2);
}

__global__ void bn_finalize_kernel(const float* __restrict__ sums,
                                   const float* __restrict__ sumsq,
                                   const float* __restrict__ gamma,
                                   const float* __restrict__ beta,
                                   float* __restrict__ scale,
                                   float* __restrict__ shift) {
    int c = threadIdx.x;          // 128 threads
    const float invN = 1.0f / (4.0f * (float)V_NODES);
    float mean = sums[c] * invN;
    float var  = sumsq[c] * invN - mean * mean;
    float sc = gamma[c] * rsqrtf(var + 1e-5f);
    scale[c] = sc;
    shift[c] = beta[c] - mean * sc;
}

__global__ __launch_bounds__(256) void bn_apply_kernel(const float* __restrict__ h,
                                                       const float* __restrict__ scale,
                                                       const float* __restrict__ shift,
                                                       float* __restrict__ y0, long n) {
    long i = (long)blockIdx.x * blockDim.x + threadIdx.x;
    long s = (long)gridDim.x * blockDim.x;
    for (; i < n; i += s) {
        int c = (int)(i & 127);
        y0[i] = h[i] * scale[c] + shift[c];
    }
}

// ------------------------------------------------------------ WMMA GEMM
// out[(v,b), o] = sum_{k,i} Xk[k][v*(4*CIN)+b*CIN+i] * W[(k*CIN+i)*COUT+o]
// MODE 0 (conv1): out = relu(gemm + bias) stored as H[v*(4*COUT)+b*COUT+o]
// MODE 1 (conv2): out = relu(gemm + bias + resid) stored as (B,COUT,V)
template <int CIN, int COUT, int MODE>
__global__ __launch_bounds__(256)
void cheb_gemm_kernel(const float* __restrict__ xk, long strideK,
                      const float* __restrict__ W,
                      const float* __restrict__ bias,
                      float* __restrict__ out,
                      const float* __restrict__ resid) {
    constexpr int KT = (4 * CIN) / 32;   // 8 (conv1) / 16 (conv2)
    constexpr int NT = COUT / 16;        // 8 (conv1) / 4  (conv2)
    static_assert(KT * NT == 64, "LDS fragment count");

    // W staged into LDS as pre-swizzled bf16 B-fragments: [frag][lane][16]
    __shared__ __attribute__((aligned(32))) __bf16 wlds[64 * 512];
    for (int t = threadIdx.x; t < 64 * 512; t += 256) {
        int fi = t >> 9, within = t & 511;
        int lane = within >> 4, e = within & 15;
        int ktile = fi / NT, nt = fi - ktile * NT;
        // B layout: lane L: col N = L%16, K = (e<8?0:16) + 8*(L/16) + e%8
        int kk = ktile * 32 + ((e & 8) << 1) + ((lane & 16) >> 1) + (e & 7);
        int o  = nt * 16 + (lane & 15);
        wlds[t] = (__bf16)W[kk * COUT + o];
    }
    __syncthreads();

    const int l    = threadIdx.x & 31;
    const int wv   = threadIdx.x >> 5;   // 8 waves/block
    const int n    = l & 15;
    const int half = l >> 4;

    const int numMTiles = (V_NODES * 4) / 16;   // 25000
    for (int mtile = blockIdx.x * 8 + wv; mtile < numMTiles; mtile += gridDim.x * 8) {
        v8f acc[NT];
        #pragma unroll
        for (int i = 0; i < NT; ++i)
            acc[i] = (v8f){0.f, 0.f, 0.f, 0.f, 0.f, 0.f, 0.f, 0.f};

        int r = mtile * 16 + n;               // A row (v,b): all lanes' row = l%16
        int v = r >> 2, b = r & 3;
        const float* arowbase = xk + (long)v * (4 * CIN) + (long)b * CIN;

        #pragma unroll 1   // rolled K loop keeps register pressure bounded (no spills)
        for (int ktile = 0; ktile < KT; ++ktile) {
            int kkbase = ktile * 32;
            int k = kkbase / CIN;
            int ibase = (kkbase - k * CIN) + 8 * half;
            const float* ap = arowbase + (long)k * strideK + ibase;
            // A layout: lane L: row M=L%16, K = (e<8?0:16) + 8*(L/16) + e%8
            float4 q0 = ((const float4*)ap)[0];   // i .. i+3
            float4 q1 = ((const float4*)ap)[1];   // i+4 .. i+7
            float4 q2 = ((const float4*)ap)[4];   // i+16 .. i+19
            float4 q3 = ((const float4*)ap)[5];   // i+20 .. i+23
            v16bf a;
            a[0]=(__bf16)q0.x; a[1]=(__bf16)q0.y; a[2]=(__bf16)q0.z; a[3]=(__bf16)q0.w;
            a[4]=(__bf16)q1.x; a[5]=(__bf16)q1.y; a[6]=(__bf16)q1.z; a[7]=(__bf16)q1.w;
            a[8]=(__bf16)q2.x; a[9]=(__bf16)q2.y; a[10]=(__bf16)q2.z; a[11]=(__bf16)q2.w;
            a[12]=(__bf16)q3.x; a[13]=(__bf16)q3.y; a[14]=(__bf16)q3.z; a[15]=(__bf16)q3.w;

            #pragma unroll
            for (int nt = 0; nt < NT; ++nt) {
                // direct __shared__ indexing -> addrspace(3) -> ds_load_b128
                v16bf bf = *(const v16bf*)(&wlds[(ktile * NT + nt) * 512 + l * 16]);
                acc[nt] = __builtin_amdgcn_wmma_f32_16x16x32_bf16(
                    false, a, false, bf, (short)0, acc[nt], false, false);
            }
            __builtin_amdgcn_sched_barrier(0);   // don't float next iter's loads up
        }

        // Epilogue. D layout: lane L: N=L%16, M = j + 8*(L/16)
        #pragma unroll
        for (int nt = 0; nt < NT; ++nt) {
            int o = nt * 16 + n;
            float bv = bias[o];
            #pragma unroll
            for (int j = 0; j < 8; ++j) {
                int m  = j + 8 * half;
                int rr = mtile * 16 + m;
                int vv = rr >> 2, bb = rr & 3;
                float val = acc[nt][j] + bv;
                if constexpr (MODE == 0) {
                    val = fmaxf(val, 0.0f);
                    out[(long)vv * (4 * COUT) + bb * COUT + o] = val;
                } else {
                    long idx = ((long)bb * COUT + o) * V_NODES + vv;
                    val += resid[idx];
                    out[idx] = fmaxf(val, 0.0f);
                }
            }
        }
    }
}

// ---------------------------------------------------------------- launch
extern "C" void kernel_launch(void* const* d_in, const int* in_sizes, int n_in,
                              void* d_out, int out_size, void* d_ws, size_t ws_size,
                              hipStream_t stream) {
    const float* x     = (const float*)d_in[0];
    const int*   rows  = (const int*)  d_in[1];
    const int*   cols  = (const int*)  d_in[2];
    const float* vals  = (const float*)d_in[3];
    const float* w1    = (const float*)d_in[4];
    const float* b1    = (const float*)d_in[5];
    const float* gamma = (const float*)d_in[6];
    const float* beta  = (const float*)d_in[7];
    const float* w2    = (const float*)d_in[8];
    const float* b2    = (const float*)d_in[9];
    float* out = (float*)d_out;

    const long S1 = (long)V_NODES * 256;   // 25,600,000
    const long S2 = (long)V_NODES * 512;   // 51,200,000
    float* ws = (float*)d_ws;
    float* X0 = ws;               // conv1 cheb levels
    float* X1 = ws + S1;
    float* X2 = ws + 2 * S1;
    float* X3 = ws + 3 * S1;
    float* H  = ws + 4 * S1;      // conv1 output (V, B*128)
    float* Y0 = ws;               // aliases X0/X1 (dead after conv1 gemm)
    float* Y1 = ws + S2;          // aliases X2/X3
    float* Y2 = ws + 2 * S2;      // aliases H (dead after bn_apply)
    float* Y3 = ws + 3 * S2;
    float* BN = ws + 4 * S2;      // 512 floats: sums, sumsq, scale, shift

    // ---- conv1: x0 = transpose(x); Chebyshev recurrence via init + scatter
    transpose_kernel<<<dim3(V_NODES / 32, 8), dim3(32, 8), 0, stream>>>(x, X0);

    fill_zero_kernel<<<2048, 256, 0, stream>>>(X1, S1);
    spmm_scatter<<<8192, 256, 0, stream>>>(rows, cols, vals, X0, X1, NNZ_E, 256, 1.0f);
    neg_copy_kernel<<<2048, 256, 0, stream>>>(X2, X0, S1);
    spmm_scatter<<<8192, 256, 0, stream>>>(rows, cols, vals, X1, X2, NNZ_E, 256, 2.0f);
    neg_copy_kernel<<<2048, 256, 0, stream>>>(X3, X1, S1);
    spmm_scatter<<<8192, 256, 0, stream>>>(rows, cols, vals, X2, X3, NNZ_E, 256, 2.0f);

    // ---- conv1 GEMM (WMMA bf16): H = relu(Xk @ W1 + b1), layout (V, B*128)
    cheb_gemm_kernel<64, 128, 0><<<3125, 256, 0, stream>>>(X0, S1, w1, b1, H, nullptr);

    // ---- BatchNorm (training stats over (B,V)) fused into conv2 input Y0
    fill_zero_kernel<<<2, 256, 0, stream>>>(BN, 512);
    bn_stats_kernel<<<2048, 256, 0, stream>>>(H, BN, BN + 128, S2);
    bn_finalize_kernel<<<1, 128, 0, stream>>>(BN, BN + 128, gamma, beta, BN + 256, BN + 384);
    bn_apply_kernel<<<2048, 256, 0, stream>>>(H, BN + 256, BN + 384, Y0, S2);

    // ---- conv2 Chebyshev levels
    fill_zero_kernel<<<2048, 256, 0, stream>>>(Y1, S2);
    spmm_scatter<<<8192, 256, 0, stream>>>(rows, cols, vals, Y0, Y1, NNZ_E, 512, 1.0f);
    neg_copy_kernel<<<2048, 256, 0, stream>>>(Y2, Y0, S2);
    spmm_scatter<<<8192, 256, 0, stream>>>(rows, cols, vals, Y1, Y2, NNZ_E, 512, 2.0f);
    neg_copy_kernel<<<2048, 256, 0, stream>>>(Y3, Y1, S2);
    spmm_scatter<<<8192, 256, 0, stream>>>(rows, cols, vals, Y2, Y3, NNZ_E, 512, 2.0f);

    // ---- conv2 GEMM + bias + residual + relu -> d_out (B,64,V)
    cheb_gemm_kernel<128, 64, 1><<<3125, 256, 0, stream>>>(Y0, S2, w2, b2, out, x);
}
